// MultiheadGAT_13460427506084
// MI455X (gfx1250) — matrix operand order repaired and checked by reading, not compiled, not run on hardware
//
#include <hip/hip_runtime.h>
#include <hip/hip_bf16.h>
#include <math.h>

// ---------------------------------------------------------------------------
// MultiheadGAT on MI455X (gfx1250, wave32).
// GEMMs use v_wmma_f32_16x16x32_bf16 (bf16 inputs, f32 accumulate),
// 32x64 output tile per wave (8 WMMAs per 32-deep K step, 1 load/WMMA).
// Edge softmax/aggregation kept in f32, L2-resident atomics.
// ---------------------------------------------------------------------------

#define Dh   192     // H*C
#define Hh   3
#define Cc   64

typedef __attribute__((ext_vector_type(16))) __bf16 v16bf;
typedef __attribute__((ext_vector_type(8)))  __bf16 v8bf;
typedef __attribute__((ext_vector_type(8)))  float  v8f;

union AV { v16bf v; v8bf h[2]; };

static __device__ __forceinline__ unsigned short f32_to_bf16(float f) {
    unsigned u = __float_as_uint(f);
    unsigned rnd = 0x7FFFu + ((u >> 16) & 1u);   // round-to-nearest-even
    return (unsigned short)((u + rnd) >> 16);
}

static __device__ __forceinline__ void atomicMaxF32(float* addr, float v) {
    // sign-aware integer punning; buffer initialized to -inf (0xFF800000)
    if (v >= 0.0f) atomicMax((int*)addr, __float_as_int(v));
    else           atomicMin((unsigned int*)addr, __float_as_uint(v));
}

// -------------------------------- fills ------------------------------------
__global__ void fillf_kernel(float* __restrict__ p, float v, int n) {
    int i = blockIdx.x * blockDim.x + threadIdx.x;
    if (i < n) p[i] = v;
}

// ------------------------- f32 -> bf16 conversion --------------------------
__global__ void cvt_bf16_kernel(const float* __restrict__ in,
                                unsigned short* __restrict__ out, int n) {
    int i = blockIdx.x * blockDim.x + threadIdx.x;
    if (i < n) out[i] = f32_to_bf16(in[i]);
}

// ------------------- transpose weights: [K,Ncol] -> bf16 [Ncol,K] ----------
__global__ void transpose_bf16_kernel(const float* __restrict__ W,
                                      unsigned short* __restrict__ Wt,
                                      int K, int Ncol) {
    int i = blockIdx.x * blockDim.x + threadIdx.x;
    if (i >= K * Ncol) return;
    int k = i % K, n = i / K;
    Wt[(size_t)n * K + k] = f32_to_bf16(W[(size_t)k * Ncol + n]);
}

// ------------------------------ WMMA GEMM ----------------------------------
// C[M,Ncol] = A[M,K](bf16, row major) x Bt[Ncol,K](bf16, row major == B^T)
// One wave computes a 32x64 output tile: per 32-deep K step it loads
// 2 A fragments (2x 16B each) + 4 B fragments (32B each) and issues
// 8x v_wmma_f32_16x16x32_bf16 -> 1 global load per WMMA, B reused across
// both M-subtiles, A reused across the 4 N-subtiles.
// Fragment layouts (contiguous k order, same in A and B; 16-bit A: lanes 0-15 hold
// K=0..15; lanes 16-31 hold K=16..31; B: lane&15 = column,
// lane>>4 selects K half; C/D: VGPR v holds rows v and v+8).
__global__ __launch_bounds__(256)
void gemm_bf16_wmma_kernel(const unsigned short* __restrict__ A,
                           const unsigned short* __restrict__ Bt,
                           float* __restrict__ C,
                           const float* __restrict__ bias,   // [Ncol] or null
                           int M, int K, int Ncol) {
    int wave = (blockIdx.x * blockDim.x + threadIdx.x) >> 5;
    int lane = threadIdx.x & 31;
    int mtiles = (M + 31) >> 5;             // 32-row tiles
    int ntiles = Ncol >> 6;                 // Ncol multiple of 64
    if (wave >= mtiles * ntiles) return;
    int tm = (wave % mtiles) << 5;
    int tn = (wave / mtiles) << 6;
    int half = lane >> 4;
    int r    = lane & 15;

    int ar0 = tm + r;       if (ar0 >= M) ar0 = M - 1;   // clamp for ragged last tile
    int ar1 = tm + 16 + r;  if (ar1 >= M) ar1 = M - 1;
    const unsigned short* arow0 = A  + (size_t)ar0 * K;
    const unsigned short* arow1 = A  + (size_t)ar1 * K;
    const unsigned short* brow  = Bt + (size_t)(tn + r) * K;  // column tn+r of B
    const size_t bstep = (size_t)16 * K;                      // 16 columns

    v8f acc[2][4];
#pragma unroll
    for (int m = 0; m < 2; ++m)
#pragma unroll
        for (int j = 0; j < 4; ++j)
            acc[m][j] = (v8f){0.f,0.f,0.f,0.f,0.f,0.f,0.f,0.f};

    for (int k0 = 0; k0 < K; k0 += 32) {
        AV a0, a1;
        a0.h[0] = *(const v8bf*)(arow0 + k0 + 16 * half);
        a0.h[1] = *(const v8bf*)(arow0 + k0 + 16 * half + 8);
        a1.h[0] = *(const v8bf*)(arow1 + k0 + 16 * half);
        a1.h[1] = *(const v8bf*)(arow1 + k0 + 16 * half + 8);
        int kb = k0 + 16 * half;
        v16bf b0 = *(const v16bf*)(brow + kb);
        v16bf b1 = *(const v16bf*)(brow + bstep     + kb);
        v16bf b2 = *(const v16bf*)(brow + bstep * 2 + kb);
        v16bf b3 = *(const v16bf*)(brow + bstep * 3 + kb);
        acc[0][0] = __builtin_amdgcn_wmma_f32_16x16x32_bf16(false, a0.v, false, b0,
                                                            (short)0, acc[0][0], false, false);
        acc[0][1] = __builtin_amdgcn_wmma_f32_16x16x32_bf16(false, a0.v, false, b1,
                                                            (short)0, acc[0][1], false, false);
        acc[0][2] = __builtin_amdgcn_wmma_f32_16x16x32_bf16(false, a0.v, false, b2,
                                                            (short)0, acc[0][2], false, false);
        acc[0][3] = __builtin_amdgcn_wmma_f32_16x16x32_bf16(false, a0.v, false, b3,
                                                            (short)0, acc[0][3], false, false);
        acc[1][0] = __builtin_amdgcn_wmma_f32_16x16x32_bf16(false, a1.v, false, b0,
                                                            (short)0, acc[1][0], false, false);
        acc[1][1] = __builtin_amdgcn_wmma_f32_16x16x32_bf16(false, a1.v, false, b1,
                                                            (short)0, acc[1][1], false, false);
        acc[1][2] = __builtin_amdgcn_wmma_f32_16x16x32_bf16(false, a1.v, false, b2,
                                                            (short)0, acc[1][2], false, false);
        acc[1][3] = __builtin_amdgcn_wmma_f32_16x16x32_bf16(false, a1.v, false, b3,
                                                            (short)0, acc[1][3], false, false);
    }

#pragma unroll
    for (int m = 0; m < 2; ++m) {
#pragma unroll
        for (int j = 0; j < 4; ++j) {
            int col = tn + j * 16 + r;
            float badd = bias ? bias[col] : 0.0f;
#pragma unroll
            for (int v = 0; v < 8; ++v) {
                int row = tm + m * 16 + v + 8 * half;
                if (row < M) C[(size_t)row * Ncol + col] = acc[m][j][v] + badd;
            }
        }
    }
}

// -------------------- attention scores a_s, a_d : [N,H] --------------------
__global__ void attn_scores_kernel(const float* __restrict__ xW,
                                   const float* __restrict__ att_s,
                                   const float* __restrict__ att_d,
                                   float* __restrict__ a_s,
                                   float* __restrict__ a_d, int Nn) {
    int i = blockIdx.x * blockDim.x + threadIdx.x;
    if (i >= Nn * Hh) return;
    int n = i / Hh, h = i % Hh;
    const float* xr = xW + (size_t)n * Dh + h * Cc;
    const float* as = att_s + h * Cc;
    const float* ad = att_d + h * Cc;
    float s = 0.f, d = 0.f;
#pragma unroll 8
    for (int c = 0; c < Cc; ++c) { float v = xr[c]; s += v * as[c]; d += v * ad[c]; }
    a_s[i] = s; a_d[i] = d;
}

// ------------------ per-edge leaky-relu score + segment max ----------------
__global__ void edge_max_kernel(const float* __restrict__ a_s,
                                const float* __restrict__ a_d,
                                const int* __restrict__ srcI,
                                const int* __restrict__ dstI,
                                float* __restrict__ ev,
                                float* __restrict__ mbuf, int Ee) {
    int e = blockIdx.x * blockDim.x + threadIdx.x;
    if (e >= Ee) return;
    int s = srcI[e], d = dstI[e];
#pragma unroll
    for (int h = 0; h < Hh; ++h) {
        float v = a_s[s * Hh + h] + a_d[d * Hh + h];
        v = v > 0.f ? v : 0.2f * v;                 // leaky_relu(0.2)
        ev[(size_t)e * Hh + h] = v;
        atomicMaxF32(&mbuf[d * Hh + h], v);
    }
}

// ----------------------- exp(e - m[dst]) + segment sum ---------------------
__global__ void edge_expsum_kernel(float* __restrict__ ev,     // in: e, out: w
                                   const float* __restrict__ mbuf,
                                   const int* __restrict__ dstI,
                                   float* __restrict__ den, int Ee) {
    int e = blockIdx.x * blockDim.x + threadIdx.x;
    if (e >= Ee) return;
    int d = dstI[e];
#pragma unroll
    for (int h = 0; h < Hh; ++h) {
        float w = __expf(ev[(size_t)e * Hh + h] - mbuf[d * Hh + h]);
        ev[(size_t)e * Hh + h] = w;
        atomicAdd(&den[d * Hh + h], w);
    }
}

// ------------- out[dst] += x[src] * alpha  (one wave per edge) -------------
__global__ __launch_bounds__(256)
void edge_aggregate_kernel(const float* __restrict__ wbuf,
                           const float* __restrict__ den,
                           const float* __restrict__ xW,
                           const int* __restrict__ srcI,
                           const int* __restrict__ dstI,
                           float* __restrict__ agg, int Ee) {
    int gw   = (blockIdx.x * blockDim.x + threadIdx.x) >> 5;
    int lane = threadIdx.x & 31;
    if (gw >= Ee) return;
    int s = srcI[gw], d = dstI[gw];
    float al0 = wbuf[(size_t)gw * Hh + 0] / (den[d * Hh + 0] + 1e-16f);
    float al1 = wbuf[(size_t)gw * Hh + 1] / (den[d * Hh + 1] + 1e-16f);
    float al2 = wbuf[(size_t)gw * Hh + 2] / (den[d * Hh + 2] + 1e-16f);
    const float* xr = xW  + (size_t)s * Dh;
    float*       ar = agg + (size_t)d * Dh;
#pragma unroll
    for (int it = 0; it < 6; ++it) {
        int c = it * 32 + lane;
        float alpha = (c < Cc) ? al0 : (c < 2 * Cc ? al1 : al2);
        atomicAdd(&ar[c], xr[c] * alpha);
    }
}

// ----------- bias + relu (in place) + column sum/sumsq partials ------------
__global__ void bias_relu_stats_kernel(float* __restrict__ agg,
                                       const float* __restrict__ bias,
                                       float* __restrict__ colsum,
                                       float* __restrict__ colsumsq,
                                       int Nn, int rowsPerBlock) {
    int c  = threadIdx.x;                       // blockDim.x == Dh (192)
    int n0 = blockIdx.x * rowsPerBlock;
    int n1 = n0 + rowsPerBlock; if (n1 > Nn) n1 = Nn;
    float b = bias[c], s = 0.f, ss = 0.f;
    for (int n = n0; n < n1; ++n) {
        size_t idx = (size_t)n * Dh + c;
        float v = agg[idx] + b;
        v = v > 0.f ? v : 0.f;
        agg[idx] = v;
        s += v; ss += v * v;
    }
    atomicAdd(&colsum[c],   s);
    atomicAdd(&colsumsq[c], ss);
}

// ------- batchnorm apply (+ optional residual add) + bf16 requantize -------
__global__ void bn_apply_kernel(const float* __restrict__ hin,
                                const float* __restrict__ colsum,
                                const float* __restrict__ colsumsq,
                                const float* __restrict__ gamma,
                                const float* __restrict__ beta,
                                float* __restrict__ h1out,        // may be null
                                const float* __restrict__ h1prev, // may be null
                                unsigned short* __restrict__ hbf,
                                int Nn, float invN) {
    int i = blockIdx.x * blockDim.x + threadIdx.x;
    if (i >= Nn * Dh) return;
    int c = i % Dh;
    float mu  = colsum[c] * invN;
    float var = colsumsq[c] * invN - mu * mu;
    float y = (hin[i] - mu) * rsqrtf(var + 1e-5f) * gamma[c] + beta[c];
    if (h1out) h1out[i] = y;
    float o = h1prev ? (h1prev[i] + y) : y;       // layer2: h_sum = h1 + h2
    hbf[i] = f32_to_bf16(o);
}

// ---------------------------------------------------------------------------
extern "C" void kernel_launch(void* const* d_in, const int* in_sizes, int n_in,
                              void* d_out, int out_size, void* d_ws, size_t ws_size,
                              hipStream_t stream) {
    const float* x   = (const float*)d_in[0];
    const float* W1  = (const float*)d_in[1];
    const float* as1 = (const float*)d_in[2];
    const float* ad1 = (const float*)d_in[3];
    const float* b1  = (const float*)d_in[4];
    const float* g1  = (const float*)d_in[5];
    const float* be1 = (const float*)d_in[6];
    const float* W2  = (const float*)d_in[7];
    const float* as2 = (const float*)d_in[8];
    const float* ad2 = (const float*)d_in[9];
    const float* b2  = (const float*)d_in[10];
    const float* g2  = (const float*)d_in[11];
    const float* be2 = (const float*)d_in[12];
    const float* Wf  = (const float*)d_in[13];
    const float* bf  = (const float*)d_in[14];
    const int*   ei  = (const int*)d_in[15];

    const int IN = 256, OUTD = 64;
    const int Nn = in_sizes[0] / IN;
    const int Ee = in_sizes[15] / 2;
    const int* srcI = ei;
    const int* dstI = ei + Ee;
    float* outp = (float*)d_out;

    // ---- scratch bump allocator (256 B aligned) ----
    char* ws = (char*)d_ws;
    size_t off = 0;
    auto alloc = [&](size_t bytes) -> void* {
        void* p = ws + off;
        off = (off + bytes + 255) & ~(size_t)255;
        return p;
    };
    unsigned short* x_bf  = (unsigned short*)alloc((size_t)Nn * IN * 2);
    unsigned short* w1t   = (unsigned short*)alloc((size_t)Dh * IN * 2);
    unsigned short* w2t   = (unsigned short*)alloc((size_t)Dh * Dh * 2);
    unsigned short* wft   = (unsigned short*)alloc((size_t)OUTD * Dh * 2);
    float* xW    = (float*)alloc((size_t)Nn * Dh * 4);
    float* a_s   = (float*)alloc((size_t)Nn * Hh * 4);
    float* a_d   = (float*)alloc((size_t)Nn * Hh * 4);
    float* mbuf  = (float*)alloc((size_t)Nn * Hh * 4);
    float* den   = (float*)alloc((size_t)Nn * Hh * 4);
    float* ew    = (float*)alloc((size_t)Ee * Hh * 4);
    float* agg   = (float*)alloc((size_t)Nn * Dh * 4);
    float* h1    = (float*)alloc((size_t)Nn * Dh * 4);
    unsigned short* hbf = (unsigned short*)alloc((size_t)Nn * Dh * 2);
    float* colsum   = (float*)alloc(Dh * 4);
    float* colsumsq = (float*)alloc(Dh * 4);
    (void)ws_size; (void)n_in; (void)out_size;

    const int TB = 256;
    auto cdiv = [](long long a, long long b) { return (int)((a + b - 1) / b); };

    // ---- prep: bf16 conversions + weight transposes ----
    cvt_bf16_kernel<<<cdiv((long long)Nn * IN, TB), TB, 0, stream>>>(x, x_bf, Nn * IN);
    transpose_bf16_kernel<<<cdiv(IN * Dh, TB), TB, 0, stream>>>(W1, w1t, IN, Dh);
    transpose_bf16_kernel<<<cdiv(Dh * Dh, TB), TB, 0, stream>>>(W2, w2t, Dh, Dh);
    transpose_bf16_kernel<<<cdiv(Dh * OUTD, TB), TB, 0, stream>>>(Wf, wft, Dh, OUTD);

    for (int layer = 0; layer < 2; ++layer) {
        // 1) projection GEMM -> xW [N, 192]
        {
            const unsigned short* A  = layer == 0 ? x_bf : hbf;
            const unsigned short* Bt = layer == 0 ? w1t : w2t;
            int K = layer == 0 ? IN : Dh;
            int waves = ((Nn + 31) / 32) * (Dh / 64);
            gemm_bf16_wmma_kernel<<<cdiv(waves, 8), TB, 0, stream>>>(
                A, Bt, xW, nullptr, Nn, K, Dh);
        }
        // 2) attention scores
        attn_scores_kernel<<<cdiv((long long)Nn * Hh, TB), TB, 0, stream>>>(
            xW, layer == 0 ? as1 : as2, layer == 0 ? ad1 : ad2, a_s, a_d, Nn);
        // 3) init reductions
        fillf_kernel<<<cdiv((long long)Nn * Hh, TB), TB, 0, stream>>>(mbuf, -INFINITY, Nn * Hh);
        fillf_kernel<<<cdiv((long long)Nn * Hh, TB), TB, 0, stream>>>(den, 0.f, Nn * Hh);
        fillf_kernel<<<cdiv((long long)Nn * Dh, TB), TB, 0, stream>>>(agg, 0.f, Nn * Dh);
        fillf_kernel<<<1, Dh, 0, stream>>>(colsum, 0.f, Dh);
        fillf_kernel<<<1, Dh, 0, stream>>>(colsumsq, 0.f, Dh);
        // 4) segment softmax over incoming edges
        edge_max_kernel<<<cdiv(Ee, TB), TB, 0, stream>>>(a_s, a_d, srcI, dstI, ew, mbuf, Ee);
        edge_expsum_kernel<<<cdiv(Ee, TB), TB, 0, stream>>>(ew, mbuf, dstI, den, Ee);
        // 5) weighted aggregation (one wave / edge)
        edge_aggregate_kernel<<<cdiv(Ee, 8), TB, 0, stream>>>(ew, den, xW, srcI, dstI, agg, Ee);
        // 6) bias + relu + BN stats
        bias_relu_stats_kernel<<<cdiv(Nn, 256), Dh, 0, stream>>>(
            agg, layer == 0 ? b1 : b2, colsum, colsumsq, Nn, 256);
        // 7) BN apply (+ residual for layer 2) + bf16 requantize for next GEMM
        bn_apply_kernel<<<cdiv((long long)Nn * Dh, TB), TB, 0, stream>>>(
            agg, colsum, colsumsq,
            layer == 0 ? g1 : g2, layer == 0 ? be1 : be2,
            layer == 0 ? h1 : nullptr,          // keep h1 (f32) for the residual
            layer == 0 ? nullptr : h1,          // layer 2: hbf = bf16(h1 + h2)
            hbf, Nn, 1.0f / (float)Nn);
    }

    // final projection: (h1 + h2) @ Wf + bf  -> d_out [N, 64] f32
    {
        int waves = ((Nn + 31) / 32) * (OUTD / 64);
        gemm_bf16_wmma_kernel<<<cdiv(waves, 8), TB, 0, stream>>>(
            hbf, wft, outp, bf, Nn, Dh, OUTD);
    }
}
